// BertSelfAttention_52226802319360
// MI455X (gfx1250) — matrix operand order repaired and checked by reading, not compiled, not run on hardware
//
#include <hip/hip_runtime.h>

typedef __attribute__((ext_vector_type(16))) __bf16 v16bf;
typedef __attribute__((ext_vector_type(8)))  float  v8f;
typedef __attribute__((ext_vector_type(4)))  unsigned int v4u;
typedef __attribute__((ext_vector_type(8)))  int v8i;
typedef __attribute__((ext_vector_type(4)))  int v4i;
typedef unsigned short us16;
typedef unsigned int   u32;
typedef unsigned long long u64;

#if __has_builtin(__builtin_amdgcn_tensor_load_to_lds) && __has_builtin(__builtin_amdgcn_s_wait_tensorcnt)
#define USE_TDM 1
#else
#define USE_TDM 0
#endif

union FragU { v16bf v; uint4 q[2]; };

__device__ __forceinline__ us16 f2bf(float x) {
  u32 u = __float_as_uint(x);
  u32 r = u + 0x7FFFu + ((u >> 16) & 1u);
  return (us16)(r >> 16);
}

// A fragment (16x32 bf16): row32 = this lane's row at its 32-element K chunk.
// lanes 0-15: K 0..7 & 16..23 ; lanes 16-31: K 8..15 & 24..31
__device__ __forceinline__ v16bf load_a_frag(const us16* row32, int half) {
  FragU f;
  const int kb = half * 8;
  f.q[0] = *(const uint4*)(row32 + kb);
  f.q[1] = *(const uint4*)(row32 + kb + 16);
  return f.v;
}

// B fragment (32x16 bf16): lane holds column n = lane&15;
// lanes 0-15: K 0..7 & 16..23 ; lanes 16-31: K 8..15 & 24..31
__device__ __forceinline__ v16bf load_b_frag(const us16* row32, int half) {
  FragU f;
  const int kb = half * 8;
  f.q[0] = *(const uint4*)(row32 + kb);
  f.q[1] = *(const uint4*)(row32 + kb + 16);
  return f.v;
}

#if USE_TDM
// D# group 0: addresses (rebuilt per tile, uniform scalar math)
__device__ __forceinline__ v4u tdm_g0(u32 lds_addr, const void* gptr) {
  const u64 ga = (u64)(size_t)gptr;
  v4u g0;
  g0.x = 1u;                                            // count=1 valid user D#
  g0.y = lds_addr;                                      // lds_addr [63:32]
  g0.z = (u32)(ga & 0xFFFFFFFFu);                       // global_addr[31:0]
  g0.w = (u32)((ga >> 32) & 0x1FFFFFFu) | (2u << 30);   // addr[56:32] | type=2
  return g0;
}

// D# group 1: dims/strides/padding (loop-invariant, hoisted)
// pad codes per D# spec: interval => 2<<code DWORDs of data, amount => (code+1) DWORDs.
__device__ __forceinline__ v8i tdm_g1(u32 tensor_d0, u32 tensor_d1,
                                      u32 tile_d0, u32 tile_d1, u32 stride0,
                                      u32 pad_int_code, u32 pad_amt_code) {
  v8i g1;
  g1[0] = (int)((1u << 16) |                            // data_size = 2 bytes
                (1u << 20) |                            // pad_enable
                (pad_int_code << 22) | (pad_amt_code << 25));
  g1[1] = (int)((tensor_d0 & 0xFFFFu) << 16);
  g1[2] = (int)(((tensor_d0 >> 16) & 0xFFFFu) | ((tensor_d1 & 0xFFFFu) << 16));
  g1[3] = (int)(((tensor_d1 >> 16) & 0xFFFFu) | ((tile_d0 & 0xFFFFu) << 16));
  g1[4] = (int)(tile_d1 & 0xFFFFu);
  g1[5] = (int)stride0;
  g1[6] = 0;
  g1[7] = 0;
  return g1;
}

__device__ __forceinline__ void tdm_issue(v4u g0, v8i g1) {
  const v4i z4 = {0, 0, 0, 0};
  const v8i z8 = {0, 0, 0, 0, 0, 0, 0, 0};
  __builtin_amdgcn_tensor_load_to_lds(g0, g1, z4, z4, z8, 0);
}
#endif

__global__ void cvt_f32_bf16x4(const float4* __restrict__ in, uint2* __restrict__ out, int n4) {
  int i = blockIdx.x * blockDim.x + threadIdx.x;
  if (i < n4) {
    float4 f = in[i];
    uint2 r;
    r.x = (u32)f2bf(f.x) | ((u32)f2bf(f.y) << 16);
    r.y = (u32)f2bf(f.z) | ((u32)f2bf(f.w) << 16);
    out[i] = r;
  }
}

// ---------------- QKV projection: C = X(4096x1024) * W^T + bias --------------
// Q,K written as [B,H,S,DH]; V written transposed as [B,H,DH,S].
__global__ __launch_bounds__(256)
void qkv_gemm(const us16* __restrict__ X,
              const us16* __restrict__ Wq, const us16* __restrict__ Wk, const us16* __restrict__ Wv,
              const float* __restrict__ bq, const float* __restrict__ bk, const float* __restrict__ bv,
              us16* __restrict__ Q, us16* __restrict__ K, us16* __restrict__ V)
{
  __shared__ us16 As[2][128][40];   // 128 rows x 32 bf16 data, 80B stride
  __shared__ us16 Ws[2][128][40];

  const int tid = threadIdx.x;
  const int wid = tid >> 5, lane = tid & 31, lanem = lane & 15, half = lane >> 4;
  const int m0 = (blockIdx.x & 31) * 128;
  const int n0 = (blockIdx.x >> 5) * 128;

  const us16* W; const float* bias; us16* Out;
  if (blockIdx.y == 0)      { W = Wq; bias = bq; Out = Q; }
  else if (blockIdx.y == 1) { W = Wk; bias = bk; Out = K; }
  else                      { W = Wv; bias = bv; Out = V; }
  const int vmode = (blockIdx.y == 2);

  const int mw = (wid >> 1) * 32;
  const int nw = (wid & 1) * 64;

  v8f z = {};
  v8f acc[2][4];
  for (int i = 0; i < 2; ++i)
    for (int j = 0; j < 4; ++j) acc[i][j] = z;

  const us16* Xrow = X + (size_t)m0 * 1024;
  const us16* Wrow = W + (size_t)n0 * 1024;

#if USE_TDM
  const u32 asAddr0 = (u32)(size_t)&As[0][0][0];
  const u32 asAddr1 = (u32)(size_t)&As[1][0][0];
  const u32 wsAddr0 = (u32)(size_t)&Ws[0][0][0];
  const u32 wsAddr1 = (u32)(size_t)&Ws[1][0][0];
  const v8i g1g = tdm_g1(1024, 4096, 32, 128, 1024, 3, 3);   // 64B rows + 16B pad
  // prologue: stage k-chunk 0 into buffer 0 (descriptors built uniformly)
  {
    const v4u g0x = tdm_g0(asAddr0, Xrow);
    const v4u g0w = tdm_g0(wsAddr0, Wrow);
    if (wid == 0) { tdm_issue(g0x, g1g); tdm_issue(g0w, g1g); }
  }
#else
  const int lrow = tid >> 1;
  const int lcol = (tid & 1) * 16;
#endif

  for (int kc = 0; kc < 32; ++kc) {
    const int cur = kc & 1;
#if USE_TDM
    {
      // next-tile descriptors computed in uniform control flow (SALU)
      const int kn = (kc + 1) * 32;
      const v4u g0x = tdm_g0(cur ? asAddr0 : asAddr1, Xrow + kn);
      const v4u g0w = tdm_g0(cur ? wsAddr0 : wsAddr1, Wrow + kn);
      if (wid == 0) {
        if (kc < 31) {
          tdm_issue(g0x, g1g);
          tdm_issue(g0w, g1g);
          __builtin_amdgcn_s_wait_tensorcnt(2);   // current pair landed; next in flight
        } else {
          __builtin_amdgcn_s_wait_tensorcnt(0);
        }
      }
    }
#else
    {
      const int k0 = kc * 32;
      const uint4* gx = (const uint4*)(Xrow + (size_t)lrow * 1024 + k0 + lcol);
      uint4* lx = (uint4*)(&As[cur][lrow][lcol]);
      lx[0] = gx[0]; lx[1] = gx[1];
      const uint4* gw = (const uint4*)(Wrow + (size_t)lrow * 1024 + k0 + lcol);
      uint4* lw = (uint4*)(&Ws[cur][lrow][lcol]);
      lw[0] = gw[0]; lw[1] = gw[1];
    }
#endif
    __syncthreads();

    v16bf af0 = load_a_frag(&As[cur][mw + 0  + lanem][0], half);
    v16bf af1 = load_a_frag(&As[cur][mw + 16 + lanem][0], half);
    for (int nt = 0; nt < 4; ++nt) {
      v16bf bf = load_b_frag(&Ws[cur][nw + nt * 16 + lanem][0], half);
      acc[0][nt] = __builtin_amdgcn_wmma_f32_16x16x32_bf16(false, af0, false, bf, (short)0, acc[0][nt], false, false);
      acc[1][nt] = __builtin_amdgcn_wmma_f32_16x16x32_bf16(false, af1, false, bf, (short)0, acc[1][nt], false, false);
    }
    __syncthreads();
  }

  for (int nt = 0; nt < 4; ++nt) {
    const int j = n0 + nw + nt * 16 + lanem;
    const float bb = bias[j];
    const int hh = j >> 6, dh = j & 63;
    for (int mt = 0; mt < 2; ++mt) {
      for (int r = 0; r < 8; ++r) {
        const int row = m0 + mw + mt * 16 + r + 8 * half;   // = s*4+b
        const int s = row >> 2, b = row & 3;
        const float val = acc[mt][nt][r] + bb;
        const size_t base = (size_t)(b * 16 + hh);
        const size_t idx = vmode ? (base * 64 + dh) * 1024 + s
                                 : (base * 1024 + s) * 64 + dh;
        Out[idx] = f2bf(val);
      }
    }
  }
}

// ---------------- Flash attention: per (b,h), 128 queries per block ----------
// K is [B,H,S,DH]; V is transposed [B,H,DH,S] so both tiles TDM straight in.
__global__ __launch_bounds__(256)
void attn(const us16* __restrict__ Q, const us16* __restrict__ K, const us16* __restrict__ Vt,
          const float* __restrict__ mask, float* __restrict__ out)
{
  __shared__ us16 Ks[2][64][72];     // 64 rows x 64 bf16 data, 144B stride
  __shared__ us16 Vts[2][64][72];    // V^T tile [dh][t]
  __shared__ us16 Ps[8][16][72];     // per-wave probs scratch 16x64

  const int tid = threadIdx.x;
  const int wid = tid >> 5, lane = tid & 31, lanem = lane & 15, half = lane >> 4;
  const int bh = blockIdx.y;
  const int b = bh >> 4, h = bh & 15;
  const int q0 = blockIdx.x * 128 + wid * 16;

  const us16* Qbh = Q + (size_t)bh * 1024 * 64;
  const us16* Kbh = K + (size_t)bh * 1024 * 64;
  const us16* Vbh = Vt + (size_t)bh * 64 * 1024;
  const float* mrow = mask + (size_t)b * 1024;

  v16bf qf0, qf1;
  {
    const us16* qr = Qbh + (size_t)(q0 + lanem) * 64;
    qf0 = load_a_frag(qr, half);
    qf1 = load_a_frag(qr + 32, half);
  }

  float rmax[8], rsum[8];
  v8f z = {};
  v8f oacc[4];
  for (int r = 0; r < 8; ++r) { rmax[r] = -3.0e38f; rsum[r] = 0.f; }
  for (int t = 0; t < 4; ++t) oacc[t] = z;

  const float scale = 0.125f;               // 1/sqrt(64)
  const float l2e = 1.44269504088896f;

#if USE_TDM
  const u32 ksAddr0 = (u32)(size_t)&Ks[0][0][0];
  const u32 ksAddr1 = (u32)(size_t)&Ks[1][0][0];
  const u32 vtAddr0 = (u32)(size_t)&Vts[0][0][0];
  const u32 vtAddr1 = (u32)(size_t)&Vts[1][0][0];
  const v8i g1k = tdm_g1(64, 1024, 64, 64, 64, 4, 3);       // 128B rows + 16B pad
  const v8i g1v = tdm_g1(1024, 64, 64, 64, 1024, 4, 3);
  {
    const v4u g0k = tdm_g0(ksAddr0, Kbh);
    const v4u g0v = tdm_g0(vtAddr0, Vbh);
    if (wid == 0) { tdm_issue(g0k, g1k); tdm_issue(g0v, g1v); }
  }
#else
  const int crow = tid >> 2;                // 0..63
  const int cq = tid & 3;
#endif

  for (int kb = 0; kb < 16; ++kb) {
    const int t0 = kb * 64;
    const int cur = kb & 1;
#if USE_TDM
    {
      const int tn = (kb + 1) * 64;
      const v4u g0k = tdm_g0(cur ? ksAddr0 : ksAddr1, Kbh + (size_t)tn * 64);
      const v4u g0v = tdm_g0(cur ? vtAddr0 : vtAddr1, Vbh + tn);
      if (wid == 0) {
        if (kb < 15) {
          tdm_issue(g0k, g1k);
          tdm_issue(g0v, g1v);
          __builtin_amdgcn_s_wait_tensorcnt(2);
        } else {
          __builtin_amdgcn_s_wait_tensorcnt(0);
        }
      }
    }
#else
    {
      const uint4* gk = (const uint4*)(Kbh + (size_t)(t0 + crow) * 64 + cq * 16);
      uint4* lk = (uint4*)(&Ks[cur][crow][cq * 16]);
      lk[0] = gk[0]; lk[1] = gk[1];
      const uint4* gv = (const uint4*)(Vbh + (size_t)crow * 1024 + t0 + cq * 16);
      uint4* lv = (uint4*)(&Vts[cur][crow][cq * 16]);
      lv[0] = gv[0]; lv[1] = gv[1];
    }
#endif
    __syncthreads();

    // scores S = Q K^T * scale + mask
    float sc[4][8];
    for (int nt = 0; nt < 4; ++nt) {
      v8f c = z;
      const us16* kr = &Ks[cur][nt * 16 + lanem][0];
      c = __builtin_amdgcn_wmma_f32_16x16x32_bf16(false, qf0, false, load_b_frag(kr, half),      (short)0, c, false, false);
      c = __builtin_amdgcn_wmma_f32_16x16x32_bf16(false, qf1, false, load_b_frag(kr + 32, half), (short)0, c, false, false);
      const float mv = mrow[t0 + nt * 16 + lanem];
      for (int r = 0; r < 8; ++r) sc[nt][r] = c[r] * scale + mv;
    }

    // online softmax: row reductions within 16-lane halves
    float bm[8];
    for (int r = 0; r < 8; ++r)
      bm[r] = fmaxf(fmaxf(sc[0][r], sc[1][r]), fmaxf(sc[2][r], sc[3][r]));
    for (int off = 8; off >= 1; off >>= 1)
      for (int r = 0; r < 8; ++r) bm[r] = fmaxf(bm[r], __shfl_xor(bm[r], off, 32));

    float corr[8], bs[8];
    for (int r = 0; r < 8; ++r) {
      const float mn = fmaxf(rmax[r], bm[r]);
      corr[r] = exp2f((rmax[r] - mn) * l2e);
      rmax[r] = mn;
      bs[r] = 0.f;
    }
    for (int nt = 0; nt < 4; ++nt)
      for (int r = 0; r < 8; ++r) {
        const float p = exp2f((sc[nt][r] - rmax[r]) * l2e);
        sc[nt][r] = p;
        bs[r] += p;
      }
    for (int off = 8; off >= 1; off >>= 1)
      for (int r = 0; r < 8; ++r) bs[r] += __shfl_xor(bs[r], off, 32);
    for (int r = 0; r < 8; ++r) rsum[r] = rsum[r] * corr[r] + bs[r];
    for (int t = 0; t < 4; ++t)
      for (int r = 0; r < 8; ++r) oacc[t][r] *= corr[r];

    // stage P (C-layout -> row-major 16x64 bf16) in per-wave LDS slice;
    // same-wave DS ops are in-order, no barrier needed.
    for (int nt = 0; nt < 4; ++nt)
      for (int r = 0; r < 8; ++r)
        Ps[wid][r + 8 * half][nt * 16 + lanem] = f2bf(sc[nt][r]);

    v16bf pf0, pf1;
    {
      const us16* pr = &Ps[wid][lanem][0];
      pf0 = load_a_frag(pr, half);
      pf1 = load_a_frag(pr + 32, half);
    }

    // O += P V
    for (int dt = 0; dt < 4; ++dt) {
      const us16* vr = &Vts[cur][dt * 16 + lanem][0];
      oacc[dt] = __builtin_amdgcn_wmma_f32_16x16x32_bf16(false, pf0, false, load_b_frag(vr, half),      (short)0, oacc[dt], false, false);
      oacc[dt] = __builtin_amdgcn_wmma_f32_16x16x32_bf16(false, pf1, false, load_b_frag(vr + 32, half), (short)0, oacc[dt], false, false);
    }
    __syncthreads();
  }

  for (int dt = 0; dt < 4; ++dt) {
    const int dh = dt * 16 + lanem;
    for (int r = 0; r < 8; ++r) {
      const int srow = q0 + r + 8 * half;
      out[(size_t)srow * 4096 + b * 1024 + h * 64 + dh] = oacc[dt][r] / rsum[r];
    }
  }
}

extern "C" void kernel_launch(void* const* d_in, const int* in_sizes, int n_in,
                              void* d_out, int out_size, void* d_ws, size_t ws_size,
                              hipStream_t stream) {
  (void)in_sizes; (void)n_in; (void)out_size; (void)ws_size;
  const float* hs   = (const float*)d_in[0];
  const float* mask = (const float*)d_in[1];
  const float* Wq   = (const float*)d_in[2];
  const float* bq   = (const float*)d_in[3];
  const float* Wk   = (const float*)d_in[4];
  const float* bk   = (const float*)d_in[5];
  const float* Wv   = (const float*)d_in[6];
  const float* bv   = (const float*)d_in[7];
  float* out = (float*)d_out;

  char* ws = (char*)d_ws;
  us16* Xb  = (us16*)(ws + 0);          // 4096x1024 bf16 : 8 MB
  us16* Wqb = (us16*)(ws + 8388608);    // 1024x1024 bf16 : 2 MB
  us16* Wkb = (us16*)(ws + 10485760);
  us16* Wvb = (us16*)(ws + 12582912);
  us16* Qb  = (us16*)(ws + 14680064);   // [B,H,S,DH]  bf16: 8 MB
  us16* Kb  = (us16*)(ws + 23068672);   // [B,H,S,DH]  bf16: 8 MB
  us16* Vtb = (us16*)(ws + 31457280);   // [B,H,DH,S]  bf16: 8 MB

  cvt_f32_bf16x4<<<(1048576 + 255) / 256, 256, 0, stream>>>((const float4*)hs, (uint2*)Xb,  1048576);
  cvt_f32_bf16x4<<<( 262144 + 255) / 256, 256, 0, stream>>>((const float4*)Wq, (uint2*)Wqb,  262144);
  cvt_f32_bf16x4<<<( 262144 + 255) / 256, 256, 0, stream>>>((const float4*)Wk, (uint2*)Wkb,  262144);
  cvt_f32_bf16x4<<<( 262144 + 255) / 256, 256, 0, stream>>>((const float4*)Wv, (uint2*)Wvb,  262144);

  qkv_gemm<<<dim3(256, 3), 256, 0, stream>>>(Xb, Wqb, Wkb, Wvb, bq, bk, bv, Qb, Kb, Vtb);
  attn<<<dim3(8, 64), 256, 0, stream>>>(Qb, Kb, Vtb, mask, out);
}
